// Vqvae_53283364274913
// MI455X (gfx1250) — matrix operand order repaired and checked by reading, not compiled, not run on hardware
//
#include <hip/hip_runtime.h>
#include <hip/hip_bf16.h>

#define NN 65536
#define KK 1024
#define DD 256
#define KC 32           // codebook rows staged in LDS per chunk
#define MTILE 128       // emb rows per block (8 waves * 16 rows)

typedef __attribute__((ext_vector_type(16))) __bf16          v16bf;
typedef __attribute__((ext_vector_type(16))) unsigned short  v16us;
typedef __attribute__((ext_vector_type(8)))  float           v8f;

static __device__ inline unsigned short f2bf_bits(float x) {
    unsigned u = __float_as_uint(x);
    unsigned r = u + 0x7FFFu + ((u >> 16) & 1u);   // round-to-nearest-even
    return (unsigned short)(r >> 16);
}
static __device__ inline float bf_bits2f(unsigned short b) {
    return __uint_as_float(((unsigned)b) << 16);
}
// monotone float -> unsigned key (ascending float == ascending uint)
static __device__ inline unsigned long long packkey(float s, int n) {
    unsigned u = __float_as_uint(s);
    u = (u & 0x80000000u) ? ~u : (u | 0x80000000u);
    return (((unsigned long long)u) << 32) | (unsigned)n;
}

__global__ void k_init(unsigned long long* minarr) {
    int i = blockIdx.x * blockDim.x + threadIdx.x;
    if (i < KK) minarr[i] = ~0ull;
}

__global__ __launch_bounds__(256) void k_main(const float* __restrict__ emb,
                                              const float* __restrict__ repr,
                                              float* __restrict__ out0,          // x_recon [N,D]
                                              unsigned long long* __restrict__ minarr) {
    __shared__ __attribute__((aligned(32))) unsigned short bh[KC * DD];  // repr hi bf16
    __shared__ __attribute__((aligned(32))) unsigned short bl[KC * DD];  // repr lo bf16
    __shared__ float              s2s[KC];                               // ||repr_k||^2
    __shared__ unsigned long long cand[KC];                              // per-k best (key|row)

    const int t    = threadIdx.x;
    const int lane = t & 31;
    const int wave = t >> 5;
    const int h    = lane >> 4;     // half-wave id (K-split of the fragment)
    const int r15  = lane & 15;
    const int m0   = blockIdx.x * MTILE + wave * 16;

    // ---- load A fragments (16 emb rows, all of D) into registers as hi/lo bf16 ----
    // A 16x32 fragment (contiguous k order): lane row=l%16 holds K = h*16+{0..15}
    const float* ap = emb + (size_t)(m0 + r15) * DD;
    v16bf a_hi[8], a_lo[8];
    float ssq = 0.f;
#pragma unroll
    for (int c = 0; c < 8; ++c) {
        const int b0 = c * 32 + h * 16;
        const int b1 = c * 32 + h * 16 + 8;
        float x[16];
        const float4 p0 = *(const float4*)(ap + b0);
        const float4 p1 = *(const float4*)(ap + b0 + 4);
        const float4 q0 = *(const float4*)(ap + b1);
        const float4 q1 = *(const float4*)(ap + b1 + 4);
        x[0]=p0.x; x[1]=p0.y; x[2]=p0.z; x[3]=p0.w;
        x[4]=p1.x; x[5]=p1.y; x[6]=p1.z; x[7]=p1.w;
        x[8]=q0.x; x[9]=q0.y; x[10]=q0.z; x[11]=q0.w;
        x[12]=q1.x; x[13]=q1.y; x[14]=q1.z; x[15]=q1.w;
        v16us th, tl;
#pragma unroll
        for (int j = 0; j < 16; ++j) {
            ssq += x[j] * x[j];
            unsigned short hb = f2bf_bits(x[j]);
            th[j] = hb;
            tl[j] = f2bf_bits(x[j] - bf_bits2f(hb));
        }
        a_hi[c] = __builtin_bit_cast(v16bf, th);
        a_lo[c] = __builtin_bit_cast(v16bf, tl);
    }
    // full ||row||^2: this lane + its h-partner cover complementary dims
    const float t2full = ssq + __shfl_xor(ssq, 16, 32);
    // t2 for the C-layout rows of this lane: row = m0 + r + 8h lives in source lane r+8h
    float t2r_[8];
#pragma unroll
    for (int r = 0; r < 8; ++r) t2r_[r] = __shfl(t2full, r + 8 * h, 32);

    float bestS[8]; int bestK[8];
#pragma unroll
    for (int r = 0; r < 8; ++r) { bestS[r] = 3.0e38f; bestK[r] = 0; }

    // ---- loop over codebook in chunks of KC ----
    for (int cb = 0; cb < KK / KC; ++cb) {
        const int kbase = cb * KC;
        // stage repr chunk to LDS as bf16 hi/lo + row norms (8 threads per row)
        {
            const int srow  = t >> 3;
            const int dbase = (t & 7) * 32;
            const float* rp = repr + (size_t)(kbase + srow) * DD + dbase;
            float sq = 0.f;
#pragma unroll
            for (int d = 0; d < 32; d += 4) {
                const float4 v = *(const float4*)(rp + d);
                float xs[4] = {v.x, v.y, v.z, v.w};
#pragma unroll
                for (int e = 0; e < 4; ++e) {
                    sq += xs[e] * xs[e];
                    unsigned short hb = f2bf_bits(xs[e]);
                    bh[srow * DD + dbase + d + e] = hb;
                    bl[srow * DD + dbase + d + e] = f2bf_bits(xs[e] - bf_bits2f(hb));
                }
            }
            sq += __shfl_xor(sq, 1, 32);
            sq += __shfl_xor(sq, 2, 32);
            sq += __shfl_xor(sq, 4, 32);
            if ((t & 7) == 0) s2s[srow] = sq;
            if (t < KC) cand[t] = ~0ull;
        }
        __syncthreads();

        // two 16-column WMMA tiles per chunk
#pragma unroll
        for (int kt = 0; kt < KC / 16; ++kt) {
            const int klb = kt * 16;
            v8f acc = {0.f, 0.f, 0.f, 0.f, 0.f, 0.f, 0.f, 0.f};
#pragma unroll
            for (int c = 0; c < 8; ++c) {
                const int boff = (klb + r15) * DD + c * 32 + h * 16;
                const v16bf bhf = *(const v16bf*)(bh + boff);
                const v16bf blf = *(const v16bf*)(bl + boff);
                acc = __builtin_amdgcn_wmma_f32_16x16x32_bf16(false, a_hi[c], false, bhf,
                                                              (short)0, acc, false, false);
                acc = __builtin_amdgcn_wmma_f32_16x16x32_bf16(false, a_hi[c], false, blf,
                                                              (short)0, acc, false, false);
                acc = __builtin_amdgcn_wmma_f32_16x16x32_bf16(false, a_lo[c], false, bhf,
                                                              (short)0, acc, false, false);
            }
            const float scol  = s2s[klb + r15];
            const int   kglob = kbase + klb + r15;
            float bs2 = 3.0e38f; int bm = 0;
#pragma unroll
            for (int r = 0; r < 8; ++r) {
                const float dot = acc[r];
                const float se = scol - 2.f * dot;               // emb-side score
                if (se < bestS[r]) { bestS[r] = se; bestK[r] = kglob; }
                const float sr = t2r_[r] - 2.f * dot;            // repr-side score
                if (sr < bs2) { bs2 = sr; bm = m0 + r + 8 * h; }
            }
            const float os = __shfl_xor(bs2, 16, 32);
            const int   om = __shfl_xor(bm, 16, 32);
            if (os < bs2) { bs2 = os; bm = om; }
            if (lane < 16) atomicMin(&cand[klb + lane], packkey(bs2, bm));
        }
        __syncthreads();
        if (t < KC) atomicMin(&minarr[kbase + t], cand[t]);
        // no extra barrier needed: cand[t] read/written only by thread t;
        // bh/bl rewritten only after the barrier above
    }

    // ---- finalize emb-side argmin across the 16-lane half, then gather x_recon ----
#pragma unroll
    for (int r = 0; r < 8; ++r) {
#pragma unroll
        for (int mask = 1; mask < 16; mask <<= 1) {
            const float os = __shfl_xor(bestS[r], mask, 32);
            const int   ok = __shfl_xor(bestK[r], mask, 32);
            if (os < bestS[r]) { bestS[r] = os; bestK[r] = ok; }
        }
        const int row = m0 + r + 8 * h;       // each half writes its own 8 rows
        const int kw  = bestK[r];             // uniform within the half
        const float4* src = (const float4*)(repr + (size_t)kw * DD);
        float4*       dst = (float4*)(out0 + (size_t)row * DD);
#pragma unroll
        for (int j = 0; j < 4; ++j) dst[r15 + 16 * j] = src[r15 + 16 * j];
    }
}

__global__ void k_copy4(const float4* __restrict__ src, float4* __restrict__ dst, int n4) {
    int i = blockIdx.x * blockDim.x + threadIdx.x;
    if (i < n4) dst[i] = src[i];
}

__global__ void k_reprx(const unsigned long long* __restrict__ minarr,
                        const float* __restrict__ emb, float* __restrict__ out3) {
    const int k = blockIdx.x;
    const int n = (int)(minarr[k] & 0xFFFFFFFFull);
    const float4* s = (const float4*)(emb + (size_t)n * DD);
    float4*       d = (float4*)(out3 + (size_t)k * DD);
    for (int j = threadIdx.x; j < DD / 4; j += blockDim.x) d[j] = s[j];
}

extern "C" void kernel_launch(void* const* d_in, const int* in_sizes, int n_in,
                              void* d_out, int out_size, void* d_ws, size_t ws_size,
                              hipStream_t stream) {
    (void)in_sizes; (void)n_in; (void)out_size; (void)ws_size;
    const float* emb  = (const float*)d_in[0];   // [N, D]
    const float* repr = (const float*)d_in[1];   // [K, D]
    float* out0 = (float*)d_out;                 // x_recon  [N,D]
    float* out1 = out0 + (size_t)NN * DD;        // emb copy [N,D]
    float* out2 = out1 + (size_t)NN * DD;        // repr copy[K,D]
    float* out3 = out2 + (size_t)KK * DD;        // repr_x   [K,D]
    unsigned long long* minarr = (unsigned long long*)d_ws;   // 8 KB

    k_init<<<(KK + 255) / 256, 256, 0, stream>>>(minarr);
    k_main<<<NN / MTILE, 256, 0, stream>>>(emb, repr, out0, minarr);

    const int n4e = NN * DD / 4;
    const int n4r = KK * DD / 4;
    k_copy4<<<(n4e + 255) / 256, 256, 0, stream>>>((const float4*)emb,  (float4*)out1, n4e);
    k_copy4<<<(n4r + 255) / 256, 256, 0, stream>>>((const float4*)repr, (float4*)out2, n4r);
    k_reprx<<<KK, 64, 0, stream>>>(minarr, emb, out3);
}
